// PagedAttention_49752901157196
// MI455X (gfx1250) — matrix operand order repaired and checked by reading, not compiled, not run on hardware
//
#include <hip/hip_runtime.h>

#define NB 2
#define SQ 2048
#define HID 2048
#define NHEAD 16
#define NCACHE 2048
#define HDIM 128
#define SKV (SQ + NCACHE)   // 4096 keys after cache concat

typedef __attribute__((ext_vector_type(16))) __bf16 v16bf;
typedef __attribute__((ext_vector_type(8)))  float  v8f;

union AFrag { v16bf v; unsigned u[8]; };
union CFrag { v8f   v; float    f[8]; };

// hardware f32->bf16 (RNE) via native conversion
__device__ __forceinline__ unsigned short f2bf(float x) {
  union { __bf16 h; unsigned short s; } u;
  u.h = (__bf16)x;
  return u.s;
}

// A-fragment K offset for 16-bit A 16x32 (ISA 7.12.2): lanes 0-15 hold K {0..7,16..23}, lanes 16-31 hold {8..15,24..31}
__device__ __forceinline__ int a_kb(int j, int g) {
  return (j < 4) ? (8 * g + 2 * j) : (16 + 8 * g + 2 * (j - 4));
}

// ---------------------------------------------------------------- prep kernels
__global__ __launch_bounds__(256) void prep_hidden(const float* __restrict__ x,
                                                   unsigned short* __restrict__ xbf) {
  size_t i = (size_t)blockIdx.x * 256 + threadIdx.x;
  xbf[i] = f2bf(x[i]);
}

// copy k/v caches into fp32 output (rows [0,NCACHE)) and bf16 workspace
__global__ __launch_bounds__(256) void prep_cache(const float* __restrict__ kc,
                                                  const float* __restrict__ vc,
                                                  float* __restrict__ koutF,
                                                  float* __restrict__ voutF,
                                                  unsigned short* __restrict__ kbf,
                                                  unsigned short* __restrict__ vbf) {
  size_t i = (size_t)blockIdx.x * 256 + threadIdx.x;   // over NB*NHEAD*NCACHE*HDIM
  int d  = (int)(i & (HDIM - 1));
  int c  = (int)((i >> 7) & (NCACHE - 1));
  int hb = (int)(i >> 18);                              // b*NHEAD + h
  size_t dst = (((size_t)hb * SKV) + c) * HDIM + d;
  float kv = kc[i], vv = vc[i];
  koutF[dst] = kv; kbf[dst] = f2bf(kv);
  voutF[dst] = vv; vbf[dst] = f2bf(vv);
}

// ---------------------------------------------------------------- GEMM: C = Xbf * W^T + bias
// X: (4096 x 2048) bf16 row-major. W: (2048 x 2048) f32 row-major (row n = output col n).
// MODE 0: Q -> bf16 ws [b][h][s][d]; MODE 1: K/V -> f32 out + bf16 ws at [b][h][CACHE+s][d]; MODE 3: plain f32 + bias.
template <int MODE>
__global__ __launch_bounds__(256)
void gemm_bf16(const unsigned short* __restrict__ X, const float* __restrict__ W,
               const float* __restrict__ bias, float* __restrict__ dstF,
               unsigned short* __restrict__ dstBf) {
  __shared__ __align__(16) unsigned short Xs[2][128 * 32];
  __shared__ __align__(16) unsigned short Wt[2][128 * 32];
  const int t = threadIdx.x;
  const int wave = t >> 5, lane = t & 31, ln = lane & 15, g = lane >> 4;
  const int wm = (wave & 3) * 32, wn = (wave >> 2) * 64;
  const int bm = blockIdx.x * 128, bn = blockIdx.y * 128;
  const int lr = t >> 1, lk = (t & 1) * 16;

  CFrag acc[2][4];
#pragma unroll
  for (int i = 0; i < 2; ++i)
#pragma unroll
    for (int jj = 0; jj < 4; ++jj)
#pragma unroll
      for (int j = 0; j < 8; ++j) acc[i][jj].f[j] = 0.0f;

  const unsigned short* xrow = X + (size_t)(bm + lr) * HID + lk;
  const float*          wrow = W + (size_t)(bn + lr) * HID + lk;

  // preload tile 0
  {
    const uint4* xg = (const uint4*)xrow;
    *(uint4*)(Xs[0] + lr * 32 + lk)     = xg[0];
    *(uint4*)(Xs[0] + lr * 32 + lk + 8) = xg[1];
    const float4* wg = (const float4*)wrow;
#pragma unroll
    for (int q4 = 0; q4 < 4; ++q4) {
      float4 w = wg[q4];
      unsigned* p = (unsigned*)(Wt[0] + lr * 32 + lk + q4 * 4);
      p[0] = (unsigned)f2bf(w.x) | ((unsigned)f2bf(w.y) << 16);
      p[1] = (unsigned)f2bf(w.z) | ((unsigned)f2bf(w.w) << 16);
    }
  }
  __syncthreads();

  for (int kt = 0; kt < HID; kt += 32) {
    const int cur = (kt >> 5) & 1, nxt = cur ^ 1;
    const bool hasNext = (kt + 32) < HID;
    // issue next tile's global loads early (registers), hide latency under WMMA
    uint4 xn0{}, xn1{};
    float4 wnx[4];
    if (hasNext) {
      const uint4* xg = (const uint4*)(xrow + kt + 32);
      xn0 = xg[0]; xn1 = xg[1];
      const float4* wg = (const float4*)(wrow + kt + 32);
#pragma unroll
      for (int q4 = 0; q4 < 4; ++q4) wnx[q4] = wg[q4];
    }

    AFrag a[2], bfr[4];
#pragma unroll
    for (int i = 0; i < 2; ++i) {
      int m = wm + i * 16 + ln;
#pragma unroll
      for (int j = 0; j < 8; ++j)
        a[i].u[j] = *(const unsigned*)(Xs[cur] + m * 32 + a_kb(j, g));
    }
#pragma unroll
    for (int jj = 0; jj < 4; ++jj) {
      int n = wn + jj * 16 + ln;
#pragma unroll
      for (int j = 0; j < 8; ++j)
        bfr[jj].u[j] = *(const unsigned*)(Wt[cur] + n * 32 + a_kb(j, g));
    }
#pragma unroll
    for (int i = 0; i < 2; ++i)
#pragma unroll
      for (int jj = 0; jj < 4; ++jj)
        acc[i][jj].v = __builtin_amdgcn_wmma_f32_16x16x32_bf16(
            false, a[i].v, false, bfr[jj].v, (short)0, acc[i][jj].v, false, false);

    if (hasNext) {
      *(uint4*)(Xs[nxt] + lr * 32 + lk)     = xn0;
      *(uint4*)(Xs[nxt] + lr * 32 + lk + 8) = xn1;
#pragma unroll
      for (int q4 = 0; q4 < 4; ++q4) {
        unsigned* p = (unsigned*)(Wt[nxt] + lr * 32 + lk + q4 * 4);
        p[0] = (unsigned)f2bf(wnx[q4].x) | ((unsigned)f2bf(wnx[q4].y) << 16);
        p[1] = (unsigned)f2bf(wnx[q4].z) | ((unsigned)f2bf(wnx[q4].w) << 16);
      }
    }
    __syncthreads();
  }

  // epilogue: C layout lane->n (ln), VGPR j -> m = 8g+j  (branch-free: MODE is compile-time)
#pragma unroll
  for (int i = 0; i < 2; ++i) {
#pragma unroll
    for (int jj = 0; jj < 4; ++jj) {
      int gn = bn + wn + jj * 16 + ln;
      float bb = bias[gn];
      int head = gn >> 7, d = gn & (HDIM - 1);
#pragma unroll
      for (int j = 0; j < 8; ++j) {
        int gm = bm + wm + i * 16 + 8 * g + j;
        float val = acc[i][jj].f[j] + bb;
        int b_ = gm >> 11, s_ = gm & (SQ - 1);
        if constexpr (MODE == 3) {
          dstF[(size_t)gm * HID + gn] = val;
        } else if constexpr (MODE == 0) {
          dstBf[(((size_t)(b_ * NHEAD + head)) * SQ + s_) * HDIM + d] = f2bf(val);
        } else {
          size_t di = (((size_t)(b_ * NHEAD + head)) * SKV + (NCACHE + s_)) * HDIM + d;
          dstF[di]  = val;
          dstBf[di] = f2bf(val);
        }
      }
    }
  }
}

// ---------------------------------------------------------------- flash attention
// block = 128 threads (4 waves); each wave owns a 16-query tile; block covers 64 queries of one (b,h)
__global__ __launch_bounds__(128)
void attn_kernel(const unsigned short* __restrict__ Qbf,
                 const unsigned short* __restrict__ Kbf,
                 const unsigned short* __restrict__ Vbf,
                 const float* __restrict__ mask,
                 unsigned short* __restrict__ attnOut) {
  __shared__ __align__(16) unsigned short Ks[2][32 * 128];   // [key][hd]
  __shared__ __align__(16) unsigned short Vt[2][128 * 36];   // [hd][key] (padded stride)
  __shared__ __align__(16) unsigned short Ps[4][16 * 32];    // per-wave P staging [m][key]
  const int t = threadIdx.x, wave = t >> 5, lane = t & 31, ln = lane & 15, g = lane >> 4;
  const int bh = blockIdx.x;                  // b*NHEAD + h
  const int b_ = bh >> 4, h_ = bh & (NHEAD - 1);
  const int qbase = blockIdx.y * 64 + wave * 16;
  const size_t headKV = (size_t)bh * SKV * HDIM;
  const size_t headQ  = (size_t)bh * SQ * HDIM;
  const float scale = 0.08838834764831845f;   // 1/sqrt(128)
  const float* mrow = mask + (size_t)b_ * SQ * SKV;

  // Q fragments held in registers for the whole key loop (4 chunks of K=32)
  AFrag qf[4];
  {
    const unsigned short* qr = Qbf + headQ + (size_t)(qbase + ln) * HDIM;
#pragma unroll
    for (int kc = 0; kc < 4; ++kc)
#pragma unroll
      for (int j = 0; j < 8; ++j)
        qf[kc].u[j] = *(const unsigned*)(qr + kc * 32 + a_kb(j, g));
  }

  CFrag o[8];
  float mi[8], li[8];
#pragma unroll
  for (int nt = 0; nt < 8; ++nt)
#pragma unroll
    for (int j = 0; j < 8; ++j) o[nt].f[j] = 0.0f;
#pragma unroll
  for (int j = 0; j < 8; ++j) { mi[j] = -1e30f; li[j] = 0.0f; }

  const int key = t >> 2, hd0 = (t & 3) * 32; // cooperative-load mapping
  const unsigned short* kbase = Kbf + headKV + (size_t)key * HDIM + hd0;
  const unsigned short* vbase = Vbf + headKV + (size_t)key * HDIM + hd0;

  // preload chunk 0
  {
    const uint4* ksrc = (const uint4*)kbase;
    uint4* kd = (uint4*)(Ks[0] + key * 128 + hd0);
    kd[0] = ksrc[0]; kd[1] = ksrc[1]; kd[2] = ksrc[2]; kd[3] = ksrc[3];
    const uint4* vsrc = (const uint4*)vbase;
    union { uint4 q[4]; unsigned short s[32]; } vv;
    vv.q[0] = vsrc[0]; vv.q[1] = vsrc[1]; vv.q[2] = vsrc[2]; vv.q[3] = vsrc[3];
#pragma unroll
    for (int i = 0; i < 32; ++i) Vt[0][(hd0 + i) * 36 + key] = vv.s[i];
  }
  __syncthreads();

  for (int k0 = 0; k0 < SKV; k0 += 32) {
    const int cur = (k0 >> 5) & 1, nxt = cur ^ 1;
    const bool hasNext = (k0 + 32) < SKV;
    // issue next chunk's global loads early (K/V stream from L2)
    uint4 kn[4], vn[4];
    if (hasNext) {
      const uint4* ksrc = (const uint4*)(kbase + (size_t)(k0 + 32) * HDIM);
      const uint4* vsrc = (const uint4*)(vbase + (size_t)(k0 + 32) * HDIM);
#pragma unroll
      for (int i = 0; i < 4; ++i) { kn[i] = ksrc[i]; vn[i] = vsrc[i]; }
      if (k0 + 64 < SKV)
        __builtin_prefetch(kbase + (size_t)(k0 + 64) * HDIM, 0, 1);
    }

    // scores: Q(16x128) * K^T(128x32) -> two 16x16 accumulators
    // B-fragment loads software-pipelined 2 deep so ds_loads overlap the WMMAs
    CFrag c0, c1;
#pragma unroll
    for (int j = 0; j < 8; ++j) { c0.f[j] = 0.0f; c1.f[j] = 0.0f; }
    AFrag b0[2], b1[2];
#pragma unroll
    for (int j = 0; j < 8; ++j) {
      int kb = a_kb(j, g);
      b0[0].u[j] = *(const unsigned*)(Ks[cur] + ln * 128 + kb);
      b1[0].u[j] = *(const unsigned*)(Ks[cur] + (16 + ln) * 128 + kb);
    }
#pragma unroll
    for (int kc = 0; kc < 4; ++kc) {
      const int pb = kc & 1, nb = pb ^ 1;
      if (kc < 3) {
#pragma unroll
        for (int j = 0; j < 8; ++j) {
          int kb = (kc + 1) * 32 + a_kb(j, g);
          b0[nb].u[j] = *(const unsigned*)(Ks[cur] + ln * 128 + kb);
          b1[nb].u[j] = *(const unsigned*)(Ks[cur] + (16 + ln) * 128 + kb);
        }
      }
      c0.v = __builtin_amdgcn_wmma_f32_16x16x32_bf16(false, qf[kc].v, false, b0[pb].v, (short)0, c0.v, false, false);
      c1.v = __builtin_amdgcn_wmma_f32_16x16x32_bf16(false, qf[kc].v, false, b1[pb].v, (short)0, c1.v, false, false);
    }

    // online softmax; row m = 8g+j lives across a 16-lane half-wave
#pragma unroll
    for (int j = 0; j < 8; ++j) {
      int qr = qbase + 8 * g + j;
      const float* mr = mrow + (size_t)qr * SKV + k0;
      float s0 = c0.f[j] * scale + mr[ln];
      float s1 = c1.f[j] * scale + mr[16 + ln];
      float mx = fmaxf(s0, s1);
      mx = fmaxf(mx, __shfl_xor(mx, 1));
      mx = fmaxf(mx, __shfl_xor(mx, 2));
      mx = fmaxf(mx, __shfl_xor(mx, 4));
      mx = fmaxf(mx, __shfl_xor(mx, 8));
      float nm = fmaxf(mi[j], mx);
      float corr = __expf(mi[j] - nm);
      mi[j] = nm;
      float e0 = __expf(s0 - nm), e1 = __expf(s1 - nm);
      float rs = e0 + e1;
      rs += __shfl_xor(rs, 1);
      rs += __shfl_xor(rs, 2);
      rs += __shfl_xor(rs, 4);
      rs += __shfl_xor(rs, 8);
      li[j] = li[j] * corr + rs;
#pragma unroll
      for (int nt = 0; nt < 8; ++nt) o[nt].f[j] *= corr;
      // stage P (C layout -> LDS row-major [m][key]) for A-fragment reload
      Ps[wave][(8 * g + j) * 32 + ln]      = f2bf(e0);
      Ps[wave][(8 * g + j) * 32 + 16 + ln] = f2bf(e1);
    }
    __builtin_amdgcn_wave_barrier();  // per-wave LDS ops are in-order; keep scheduler honest

    // O += P(16x32) * V(32x128), V-fragment loads pipelined 2 deep
    AFrag pa;
#pragma unroll
    for (int j = 0; j < 8; ++j)
      pa.u[j] = *(const unsigned*)(&Ps[wave][ln * 32 + a_kb(j, g)]);
    AFrag vb[2];
#pragma unroll
    for (int j = 0; j < 8; ++j)
      vb[0].u[j] = *(const unsigned*)(Vt[cur] + ln * 36 + a_kb(j, g));
#pragma unroll
    for (int nt = 0; nt < 8; ++nt) {
      const int pb = nt & 1, nb = pb ^ 1;
      if (nt < 7) {
        int hd = (nt + 1) * 16 + ln;
#pragma unroll
        for (int j = 0; j < 8; ++j)
          vb[nb].u[j] = *(const unsigned*)(Vt[cur] + hd * 36 + a_kb(j, g));
      }
      o[nt].v = __builtin_amdgcn_wmma_f32_16x16x32_bf16(false, pa.v, false, vb[pb].v, (short)0, o[nt].v, false, false);
    }

    // commit next chunk into the other buffer
    if (hasNext) {
      uint4* kd = (uint4*)(Ks[nxt] + key * 128 + hd0);
      kd[0] = kn[0]; kd[1] = kn[1]; kd[2] = kn[2]; kd[3] = kn[3];
      union { uint4 q[4]; unsigned short s[32]; } vv;
      vv.q[0] = vn[0]; vv.q[1] = vn[1]; vv.q[2] = vn[2]; vv.q[3] = vn[3];
#pragma unroll
      for (int i = 0; i < 32; ++i) Vt[nxt][(hd0 + i) * 36 + key] = vv.s[i];
    }
    __syncthreads();
  }

  // normalize and emit attn (bf16) in [b][s][h*HD+d] layout for the O-projection GEMM
  float inv[8];
#pragma unroll
  for (int j = 0; j < 8; ++j) inv[j] = 1.0f / li[j];
#pragma unroll
  for (int nt = 0; nt < 8; ++nt) {
    int hd = nt * 16 + ln;
#pragma unroll
    for (int j = 0; j < 8; ++j) {
      int qr = qbase + 8 * g + j;
      attnOut[((size_t)(b_ * SQ + qr)) * HID + h_ * HDIM + hd] = f2bf(o[nt].f[j] * inv[j]);
    }
  }
}

// ---------------------------------------------------------------- launch
extern "C" void kernel_launch(void* const* d_in, const int* in_sizes, int n_in,
                              void* d_out, int out_size, void* d_ws, size_t ws_size,
                              hipStream_t stream) {
  (void)in_sizes; (void)n_in; (void)out_size; (void)ws_size;
  const float* hidden = (const float*)d_in[0];
  const float* mask   = (const float*)d_in[1];
  const float* kcache = (const float*)d_in[2];
  const float* vcache = (const float*)d_in[3];
  const float* Wq = (const float*)d_in[4];  const float* bq = (const float*)d_in[5];
  const float* Wk = (const float*)d_in[6];  const float* bk = (const float*)d_in[7];
  const float* Wv = (const float*)d_in[8];  const float* bv = (const float*)d_in[9];
  const float* Wo = (const float*)d_in[10]; const float* bo = (const float*)d_in[11];

  const size_t BSH = (size_t)NB * SQ * HID;            // 8,388,608
  const size_t KVN = (size_t)NB * NHEAD * SKV * HDIM;  // 16,777,216

  float* out  = (float*)d_out;
  float* Kout = out + BSH;
  float* Vout = Kout + KVN;

  unsigned short* ws  = (unsigned short*)d_ws;
  unsigned short* Xbf = ws;
  unsigned short* Qbf = Xbf + BSH;
  unsigned short* Kbf = Qbf + BSH;
  unsigned short* Vbf = Kbf + KVN;
  unsigned short* Abf = Vbf + KVN;

  prep_hidden<<<(unsigned)(BSH / 256), 256, 0, stream>>>(hidden, Xbf);
  const size_t CELEMS = (size_t)NB * NHEAD * NCACHE * HDIM;  // 8,388,608
  prep_cache<<<(unsigned)(CELEMS / 256), 256, 0, stream>>>(kcache, vcache, Kout, Vout, Kbf, Vbf);

  dim3 gg(NB * SQ / 128, HID / 128);  // (32, 16)
  gemm_bf16<0><<<gg, 256, 0, stream>>>(Xbf, Wq, bq, nullptr, Qbf);
  gemm_bf16<1><<<gg, 256, 0, stream>>>(Xbf, Wk, bk, Kout, Kbf);
  gemm_bf16<1><<<gg, 256, 0, stream>>>(Xbf, Wv, bv, Vout, Vbf);

  attn_kernel<<<dim3(NB * NHEAD, SQ / 64), 128, 0, stream>>>(Qbf, Kbf, Vbf, mask, Abf);

  gemm_bf16<3><<<gg, 256, 0, stream>>>(Abf, Wo, bo, out, nullptr);
}
